// BiLSTM_CRF_71322226917934
// MI455X (gfx1250) — compile-verified, matrix operand order checked by reading, not run on hardware
//
#include <hip/hip_runtime.h>
#include <hip/hip_bf16.h>
#include <stdint.h>

typedef __attribute__((ext_vector_type(16))) __bf16        v16bf;
typedef __attribute__((ext_vector_type(8)))  float         v8f;
typedef __attribute__((ext_vector_type(4)))  unsigned int  v4u;

union FragU { v4u u[2]; v16bf v; };

#define B_    64
#define T_    512
#define E_    300
#define EP_   320      // E padded to multiple of 32 for WMMA K
#define H_    128
#define G4_   512      // 4*H gate width
#define K_    3
#define M_TOT (B_*T_)  // 32768 rows

// ---------- helpers ----------
__device__ __forceinline__ unsigned short f2bf(float f) {
  unsigned int u = __float_as_uint(f);
  u += 0x7FFFu + ((u >> 16) & 1u);            // round-to-nearest-even
  return (unsigned short)(u >> 16);
}

__device__ __forceinline__ float sigm(float x) {
  return 1.0f / (1.0f + __expf(-x));
}

// A/B 16-bit fragment per ISA layout: lane holds row/col (lane&15), K runs
// [k0..k0+7] and [k0+16..k0+23] with k0 = 8*(lane>>4) (+32 per K-tile).
// p points at element k0 of the lane's row: two 16-byte loads.
__device__ __forceinline__ v16bf load_frag(const unsigned short* p) {
  FragU f;
  f.u[0] = *(const v4u*)(p);
  f.u[1] = *(const v4u*)(p + 16);
  return f.v;
}

// ---------- kernel 1: embedding gather + bf16 cast (K-padded) ----------
__global__ void k_embed(const int* __restrict__ tokens,
                        const float* __restrict__ emb,
                        unsigned short* __restrict__ xb) {
  int gid = blockIdx.x * blockDim.x + threadIdx.x;
  if (gid >= M_TOT * EP_) return;
  int row = gid / EP_;
  int col = gid - row * EP_;
  float v = 0.0f;
  if (col < E_) {
    int tok = tokens[row];
    v = emb[(size_t)tok * E_ + col];
  }
  xb[gid] = f2bf(v);
}

// ---------- kernel 2: transpose + cast weights to N-major bf16 ----------
// src: [K,N] f32 row-major;  dst: [N,Kpad] bf16 (zero pad K..Kpad)
__global__ void k_tcast(const float* __restrict__ src,
                        unsigned short* __restrict__ dst,
                        int K, int N, int Kpad) {
  int gid = blockIdx.x * blockDim.x + threadIdx.x;
  if (gid >= N * Kpad) return;
  int n = gid / Kpad;
  int k = gid - n * Kpad;
  float v = (k < K) ? src[(size_t)k * N + n] : 0.0f;
  dst[gid] = f2bf(v);
}

// ---------- kernel 3: input GEMM  xW = X(bf16) @ W(bf16) + b ----------
// X: [32768, 320] bf16, Wt: [512, 320] bf16 (N-major), out: [32768, 512] f32
__global__ __launch_bounds__(256) void k_ingemm(
    const unsigned short* __restrict__ xb,
    const unsigned short* __restrict__ wt,
    const float* __restrict__ bias,
    float* __restrict__ xw) {
  const int wave = threadIdx.x >> 5;
  const int lane = threadIdx.x & 31;
  const int tile = blockIdx.x * 8 + wave;     // 2048 mTiles * 32 nTiles
  const int mTile = tile >> 5;
  const int nTile = tile & 31;
  const int ln15  = lane & 15;
  const int hl    = lane >> 4;
  const int koff  = hl << 3;                  // 0 or 8

  const unsigned short* ap = xb + (size_t)((mTile << 4) + ln15) * EP_ + koff;
  const unsigned short* bptr = wt + (size_t)((nTile << 4) + ln15) * EP_ + koff;

  v8f acc = {};
#pragma unroll
  for (int kt = 0; kt < EP_ / 32; ++kt) {
    v16bf a = load_frag(ap + kt * 32);
    v16bf b = load_frag(bptr + kt * 32);
    acc = __builtin_amdgcn_wmma_f32_16x16x32_bf16(
        false, a, false, b, (short)0, acc, false, false);
  }
  const int ncol = (nTile << 4) + ln15;
  const float bv = bias[ncol];
#pragma unroll
  for (int i = 0; i < 8; ++i) {
    int m = (mTile << 4) + i + (hl << 3);     // C layout: M = i + 8*(lane>>4)
    xw[(size_t)m * G4_ + ncol] = acc[i] + bv;
  }
}

// ---------- kernel 4: recurrent LSTM scan (one workgroup per direction) ----
// LDS: U^T 512x136 bf16 (139264B) | h 64x136 bf16 (17408B) | z 64x516 f32
__global__ __launch_bounds__(1024, 1) void k_lstm(
    const float* __restrict__ xw,             // [2][32768][512] f32
    const unsigned short* __restrict__ uft,   // [512][128] bf16 N-major
    const unsigned short* __restrict__ ubt,
    float* __restrict__ hs) {                 // [2][B][T][H] f32
  extern __shared__ unsigned char smem[];
  unsigned short* ut_lds = (unsigned short*)smem;
  unsigned short* h_lds  = (unsigned short*)(smem + 512 * 136 * 2);
  float*          z_lds  = (float*)(smem + 512 * 136 * 2 + 64 * 136 * 2);

  const int dir = blockIdx.x;
  const int tid = threadIdx.x;
  const unsigned short* ut = dir ? ubt : uft;

  // Stage U^T into LDS (128 KB, resident for all 512 steps) with the
  // CDNA5 async global->LDS engine: one GLOBAL_LOAD_ASYNC_TO_LDS_B128 per
  // 16-byte chunk (8192 chunks), tracked by ASYNCcnt, fenced before the
  // workgroup barrier. Global rows are 256 B; LDS rows padded to 272 B.
  {
    unsigned ldsbase = (unsigned)(size_t)(void*)ut_lds;  // LDS byte offset
#pragma unroll
    for (int k = 0; k < 8; ++k) {
      int c   = tid + k * 1024;     // chunk id 0..8191
      int row = c >> 4;             // 16 chunks per 256B global row
      int cc  = c & 15;
      unsigned long long ga =
          (unsigned long long)(size_t)((const char*)ut + row * 256 + cc * 16);
      unsigned la = ldsbase + (unsigned)(row * 272 + cc * 16);
      asm volatile("global_load_async_to_lds_b128 %0, %1, off"
                   :: "v"(la), "v"(ga) : "memory");
    }
    asm volatile("s_wait_asynccnt 0" ::: "memory");
  }
  for (int i = tid; i < 64 * 136; i += 1024) h_lds[i] = 0;  // h0 = 0
  __syncthreads();

  const int wave = tid >> 5;
  const int lane = tid & 31;
  const int ln15 = lane & 15;
  const int hl   = lane >> 4;
  const int koff = hl << 3;

  // Hoist B-fragments of U for this wave's 16-column slab: live 512 steps.
  v16bf bfrag[4];
  {
    const unsigned short* bp0 = ut_lds + (size_t)((wave << 4) + ln15) * 136 + koff;
#pragma unroll
    for (int kt = 0; kt < 4; ++kt) bfrag[kt] = load_frag(bp0 + kt * 32);
  }

  const int gb = tid >> 4;          // batch row owned in gate phase
  const int jb = (tid & 15) << 3;   // 8 hidden units per thread
  float creg[8];
#pragma unroll
  for (int r = 0; r < 8; ++r) creg[r] = 0.0f;

  const float* xw_dir = xw + (size_t)dir * M_TOT * G4_;

  for (int s = 0; s < T_; ++s) {
    const int t = dir ? (T_ - 1 - s) : s;
    const float* xwrow = xw_dir + ((size_t)gb * T_ + t) * G4_;
    __builtin_prefetch(xwrow + jb, 0, 1);
    __builtin_prefetch(xwrow + 128 + jb, 0, 1);
    __builtin_prefetch(xwrow + 256 + jb, 0, 1);
    __builtin_prefetch(xwrow + 384 + jb, 0, 1);

    // Phase A: z = h @ U via WMMA (4 mTiles x 4 K-steps per wave)
#pragma unroll
    for (int mt = 0; mt < 4; ++mt) {
      v8f acc = {};
      const unsigned short* ap = h_lds + (size_t)((mt << 4) + ln15) * 136 + koff;
#pragma unroll
      for (int kt = 0; kt < 4; ++kt) {
        v16bf a = load_frag(ap + kt * 32);
        acc = __builtin_amdgcn_wmma_f32_16x16x32_bf16(
            false, a, false, bfrag[kt], (short)0, acc, false, false);
      }
      const int col = (wave << 4) + ln15;
#pragma unroll
      for (int i = 0; i < 8; ++i) {
        int m = (mt << 4) + i + (hl << 3);
        z_lds[m * 516 + col] = acc[i];
      }
    }
    __syncthreads();

    // Phase B: gates i,f,g,o -> c,h ; write h back to LDS (bf16) + global
    const float* zrow = z_lds + gb * 516;
    float* hsrow = hs + (((size_t)dir * B_ + gb) * T_ + t) * H_;
#pragma unroll
    for (int r = 0; r < 8; ++r) {
      int j = jb + r;
      float zi = zrow[j]       + xwrow[j];
      float zf = zrow[128 + j] + xwrow[128 + j];
      float zg = zrow[256 + j] + xwrow[256 + j];
      float zo = zrow[384 + j] + xwrow[384 + j];
      float c  = sigm(zf) * creg[r] + sigm(zi) * tanhf(zg);
      creg[r] = c;
      float h  = sigm(zo) * tanhf(c);
      h_lds[gb * 136 + j] = f2bf(h);
      hsrow[j] = h;
    }
    __syncthreads();
  }
}

// ---------- kernel 5: dense -> logits -> CRF potentials (K=3) ----------
__global__ void k_pot(const float* __restrict__ hs,
                      const float* __restrict__ Wd, const float* __restrict__ bd,
                      const float* __restrict__ crfW, const float* __restrict__ crfb,
                      const float* __restrict__ leftb, const float* __restrict__ rightb,
                      float* __restrict__ pot) {
  int gid = blockIdx.x * blockDim.x + threadIdx.x;
  if (gid >= M_TOT) return;
  int b = gid / T_;
  int t = gid - b * T_;
  const float* h0 = hs + ((size_t)b * T_ + t) * H_;
  const float* h1 = hs + (((size_t)B_ + b) * T_ + t) * H_;
  float lg0 = bd[0], lg1 = bd[1], lg2 = bd[2];
  for (int j = 0; j < H_; ++j) {
    float v = h0[j];
    lg0 += v * Wd[j * 3 + 0]; lg1 += v * Wd[j * 3 + 1]; lg2 += v * Wd[j * 3 + 2];
  }
  for (int j = 0; j < H_; ++j) {
    float v = h1[j];
    lg0 += v * Wd[(H_ + j) * 3 + 0];
    lg1 += v * Wd[(H_ + j) * 3 + 1];
    lg2 += v * Wd[(H_ + j) * 3 + 2];
  }
#pragma unroll
  for (int k = 0; k < K_; ++k) {
    float p = crfb[k] + lg0 * crfW[k] + lg1 * crfW[3 + k] + lg2 * crfW[6 + k];
    if (t == 0)      p += leftb[k];
    if (t == T_ - 1) p += rightb[k];
    pot[(size_t)gid * K_ + k] = p;
  }
}

// ---------- kernel 6: Viterbi decode (one lane per batch row) ----------
__global__ void k_viterbi(const float* __restrict__ pot,
                          const float* __restrict__ chain,
                          int* __restrict__ bp,
                          float* __restrict__ outTags) {
  int b = threadIdx.x;
  if (b >= B_) return;
  float ch[9];
#pragma unroll
  for (int i = 0; i < 9; ++i) ch[i] = chain[i];
  float a0 = pot[(size_t)b * T_ * 3 + 0];
  float a1 = pot[(size_t)b * T_ * 3 + 1];
  float a2 = pot[(size_t)b * T_ * 3 + 2];
  for (int t = 1; t < T_; ++t) {
    const float* pr = pot + ((size_t)b * T_ + t) * 3;
    float na[3]; int nb[3];
#pragma unroll
    for (int k = 0; k < 3; ++k) {
      float s0 = a0 + ch[k];
      float s1 = a1 + ch[3 + k];
      float s2 = a2 + ch[6 + k];
      int arg = 0; float best = s0;
      if (s1 > best) { best = s1; arg = 1; }   // first-index wins, like argmax
      if (s2 > best) { best = s2; arg = 2; }
      na[k] = best + pr[k];
      nb[k] = arg;
    }
    a0 = na[0]; a1 = na[1]; a2 = na[2];
    int* bpr = bp + ((size_t)b * T_ + t) * 3;
    bpr[0] = nb[0]; bpr[1] = nb[1]; bpr[2] = nb[2];
  }
  int tag = 0; float best = a0;
  if (a1 > best) { best = a1; tag = 1; }
  if (a2 > best) { best = a2; tag = 2; }
  outTags[(size_t)b * T_ + (T_ - 1)] = (float)tag;
  for (int t = T_ - 2; t >= 0; --t) {
    tag = bp[((size_t)b * T_ + (t + 1)) * 3 + tag];
    outTags[(size_t)b * T_ + t] = (float)tag;
  }
}

// ---------- host ----------
extern "C" void kernel_launch(void* const* d_in, const int* in_sizes, int n_in,
                              void* d_out, int out_size, void* d_ws, size_t ws_size,
                              hipStream_t stream) {
  (void)in_sizes; (void)n_in; (void)out_size; (void)ws_size;
  const int*   tokens = (const int*)d_in[0];
  const float* emb    = (const float*)d_in[1];
  const float* Wf     = (const float*)d_in[2];
  const float* Uf     = (const float*)d_in[3];
  const float* bf     = (const float*)d_in[4];
  const float* Wb     = (const float*)d_in[5];
  const float* Ub     = (const float*)d_in[6];
  const float* bb     = (const float*)d_in[7];
  const float* Wd     = (const float*)d_in[8];
  const float* bd     = (const float*)d_in[9];
  const float* crfW   = (const float*)d_in[10];
  const float* crfb   = (const float*)d_in[11];
  const float* chain  = (const float*)d_in[12];
  const float* leftb  = (const float*)d_in[13];
  const float* rightb = (const float*)d_in[14];

  unsigned char* ws = (unsigned char*)d_ws;
  unsigned short* xb  = (unsigned short*)(ws + 0);           // 32768*320*2  = 20971520
  unsigned short* wft = (unsigned short*)(ws + 20971520);    // 512*320*2    = 327680
  unsigned short* wbt = (unsigned short*)(ws + 21299200);    // 327680
  unsigned short* uft = (unsigned short*)(ws + 21626880);    // 512*128*2    = 131072
  unsigned short* ubt = (unsigned short*)(ws + 21757952);    // 131072
  float*          xw  = (float*)(ws + 21889024);             // 2*32768*512*4 = 134217728
  float*          hs  = (float*)(ws + 156106752);            // 2*64*512*128*4 = 33554432
  int*            bp  = (int*)(ws + 189661184);              // 64*512*3*4   = 393216

  float* outTags = (float*)d_out;            // decoded [B,T]
  float* outPot  = (float*)d_out + M_TOT;    // potentials [B,T,K]

  k_embed<<<(M_TOT * EP_) / 256, 256, 0, stream>>>(tokens, emb, xb);
  k_tcast<<<(512 * 320) / 256, 256, 0, stream>>>(Wf, wft, E_, G4_, EP_);
  k_tcast<<<(512 * 320) / 256, 256, 0, stream>>>(Wb, wbt, E_, G4_, EP_);
  k_tcast<<<(512 * 128) / 256, 256, 0, stream>>>(Uf, uft, H_, G4_, H_);
  k_tcast<<<(512 * 128) / 256, 256, 0, stream>>>(Ub, ubt, H_, G4_, H_);

  k_ingemm<<<8192, 256, 0, stream>>>(xb, wft, bf, xw);
  k_ingemm<<<8192, 256, 0, stream>>>(xb, wbt, bb, xw + (size_t)M_TOT * G4_);

  const size_t lds = 512 * 136 * 2 + 64 * 136 * 2 + 64 * 516 * 4;  // 288768 B
  k_lstm<<<2, 1024, lds, stream>>>(xw, uft, ubt, hs);

  k_pot<<<M_TOT / 256, 256, 0, stream>>>(hs, Wd, bd, crfW, crfb, leftb, rightb, outPot);
  k_viterbi<<<1, 64, 0, stream>>>(outPot, chain, bp, outTags);
}
